// GraphNeuralNetwork_12541304505018
// MI455X (gfx1250) — matrix operand order repaired and checked by reading, not compiled, not run on hardware
//
#include <hip/hip_runtime.h>
#include <hip/hip_bf16.h>
#include <math.h>

typedef __attribute__((ext_vector_type(16))) __bf16 v16bf;
typedef __attribute__((ext_vector_type(8)))  float  v8f;

#define HID 64

// ---------------------------------------------------------------------------
// Dense GEMM: out[M x 64] = H[M x KDIM] @ W[KDIM x 64], bf16 WMMA, f32 accum.
// Block = 256 threads (8 waves). Each wave computes a 16-row x 64-col strip:
// one A fragment per K-step feeds 4 WMMAs (N-tiles 0..3). W is pre-permuted
// into B-fragment order in LDS so each lane loads its fragment as one
// contiguous 32B vector (ds_load_b128 x2) instead of 16 scalar u16 loads.
// ---------------------------------------------------------------------------
template <int KDIM>
__global__ __launch_bounds__(256)
void gemm_wmma(const float* __restrict__ H, const float* __restrict__ W,
               float* __restrict__ out, int M) {
  // B-fragment order: unit index (kb*4 + tn)*32 + lane, 16 bf16 per unit.
  //   k = kb*32 + kr ; n = tn*16 + c
//   lane = ((kr >> 3) & 1) ? 16 + c : c ; element e = (kr & 7) + ((kr >> 4) << 3)
  __shared__ alignas(32) __bf16 Bp[KDIM * HID];
  for (int i = threadIdx.x; i < KDIM * HID; i += 256) {
    int k = i / HID, n = i % HID;
    int kb = k >> 5, kr = k & 31;
    int tn = n >> 4, c = n & 15;
    int lane = ((kr >> 3) & 1) ? (16 + c) : c;
    int e = (kr & 7) + ((kr >> 4) << 3);
    Bp[((((kb << 2) + tn) << 5) + lane) * 16 + e] = (__bf16)W[i];
  }
  __syncthreads();

  const int wave = threadIdx.x >> 5;
  const int lane = threadIdx.x & 31;
  const int row0 = (blockIdx.x * 8 + wave) * 16;
  if (row0 >= M) return;                      // wave-uniform

  const int l15 = lane & 15;
  const bool hi = lane >= 16;
  const int aRow = row0 + l15;

  v8f acc0 = {}, acc1 = {}, acc2 = {}, acc3 = {};
  const v16bf* Bv = reinterpret_cast<const v16bf*>(Bp);

  for (int kb = 0; kb < (KDIM >> 5); ++kb) {
    // A fragment (16x32 bf16), ISA layout:
    //   lanes 0-15 : row = lane,    K = kb*32 + {0..7, 16..23}
    //   lanes 16-31: row = lane-16, K = kb*32 + {8..15, 24..31}
    v16bf a;
    const float* hrow = H + (size_t)aRow * KDIM + (kb << 5) + (hi ? 8 : 0);
#pragma unroll
    for (int e = 0; e < 8; ++e)  a[e] = (__bf16)hrow[e];
#pragma unroll
    for (int e = 8; e < 16; ++e) a[e] = (__bf16)hrow[e + 8];

    const v16bf* bp = Bv + (kb << 7) + lane;  // + tn*32
    acc0 = __builtin_amdgcn_wmma_f32_16x16x32_bf16(false, a, false, bp[0],
                                                   (short)0, acc0, false, false);
    acc1 = __builtin_amdgcn_wmma_f32_16x16x32_bf16(false, a, false, bp[32],
                                                   (short)0, acc1, false, false);
    acc2 = __builtin_amdgcn_wmma_f32_16x16x32_bf16(false, a, false, bp[64],
                                                   (short)0, acc2, false, false);
    acc3 = __builtin_amdgcn_wmma_f32_16x16x32_bf16(false, a, false, bp[96],
                                                   (short)0, acc3, false, false);
  }

  // C/D layout: VGPR r -> M = r (lanes 0-15) / 8+r (lanes 16-31), N = lane&15
  float* obase = out + (size_t)row0 * HID + l15;
  const int mOff = hi ? 8 : 0;
#pragma unroll
  for (int r = 0; r < 8; ++r) {
    float* o = obase + (size_t)(mOff + r) * HID;
    o[0]  = acc0[r];
    o[16] = acc1[r];
    o[32] = acc2[r];
    o[48] = acc3[r];
  }
}

// ---------------------------------------------------------------------------
// Degree / norm
// ---------------------------------------------------------------------------
__global__ void deg_init(float* __restrict__ deg, int n) {
  int t = blockIdx.x * blockDim.x + threadIdx.x;
  if (t < n) deg[t] = 1.0f;               // self-loop contributes 1
}
__global__ void deg_scatter(const int* __restrict__ dst, float* __restrict__ deg, int e) {
  int t = blockIdx.x * blockDim.x + threadIdx.x;
  if (t < e) atomicAdd(&deg[dst[t]], 1.0f);
}
__global__ void deg_finalize(float* __restrict__ deg, int n) {  // deg -> dinv in place
  int t = blockIdx.x * blockDim.x + threadIdx.x;
  if (t < n) deg[t] = rsqrtf(deg[t]);
}

// ---------------------------------------------------------------------------
// Per-layer: init accumulator with bias + self-loop term, scatter edges, relu
// ---------------------------------------------------------------------------
__global__ void self_init(const float* __restrict__ hw, const float* __restrict__ dinv,
                          const float* __restrict__ bias, float* __restrict__ out, int n) {
  int t = blockIdx.x * blockDim.x + threadIdx.x;
  if (t >= n) return;
  int v = t >> 6, j = t & 63;
  float di = dinv[v];
  out[t] = bias[j] + hw[t] * di * di;
}

__global__ void edge_scatter(const int* __restrict__ src, const int* __restrict__ dst,
                             const float* __restrict__ dinv, const float* __restrict__ hw,
                             float* __restrict__ out, int nEdges) {
  int t = blockIdx.x * blockDim.x + threadIdx.x;
  int e = t >> 4;                         // 16 lanes per edge
  if (e >= nEdges) return;
  int j = (t & 15) * 4;
  int s = src[e], d = dst[e];
  float nrm = dinv[s] * dinv[d];
  const float4 v = *reinterpret_cast<const float4*>(hw + (size_t)s * HID + j);
  float* o = out + (size_t)d * HID + j;
  atomicAdd(o + 0, v.x * nrm);
  atomicAdd(o + 1, v.y * nrm);
  atomicAdd(o + 2, v.z * nrm);
  atomicAdd(o + 3, v.w * nrm);
}

__global__ void relu_inplace(float* __restrict__ x, int n) {
  int t = blockIdx.x * blockDim.x + threadIdx.x;
  if (t < n) x[t] = fmaxf(x[t], 0.0f);
}

// ---------------------------------------------------------------------------
// Pooling + classifier
// ---------------------------------------------------------------------------
__global__ void zero_f32(float* __restrict__ p, int n) {
  int t = blockIdx.x * blockDim.x + threadIdx.x;
  if (t < n) p[t] = 0.0f;
}

__global__ void pool_sum(const float* __restrict__ x1, const float* __restrict__ x2,
                         const float* __restrict__ x3, const int* __restrict__ batch,
                         float* __restrict__ pooled, float* __restrict__ cnt, int nNodes) {
  int t = blockIdx.x * blockDim.x + threadIdx.x;
  int v = t >> 6, j = t & 63;
  if (v >= nNodes) return;
  int g = batch[v];
  float* pg = pooled + (size_t)g * (3 * HID);
  size_t o = (size_t)v * HID + j;
  atomicAdd(pg + j,            x1[o]);
  atomicAdd(pg + HID + j,      x2[o]);
  atomicAdd(pg + 2 * HID + j,  x3[o]);
  if (j == 0) atomicAdd(cnt + g, 1.0f);
}

__global__ __launch_bounds__(32)
void classify(const float* __restrict__ pooled, const float* __restrict__ cnt,
              const float* __restrict__ Wl, const float* __restrict__ bl,
              float* __restrict__ out) {
  __shared__ float m[3 * HID];
  __shared__ float lg[10];
  __shared__ float red[2];
  int g = blockIdx.x;
  float c = fmaxf(cnt[g], 1.0f);
  for (int j = threadIdx.x; j < 3 * HID; j += 32)
    m[j] = pooled[(size_t)g * (3 * HID) + j] / c;
  __syncthreads();
  if (threadIdx.x < 10) {
    float acc = bl[threadIdx.x];
    for (int j = 0; j < 3 * HID; ++j)
      acc += m[j] * Wl[j * 10 + threadIdx.x];
    lg[threadIdx.x] = acc;
  }
  __syncthreads();
  if (threadIdx.x == 0) {
    float mx = lg[0];
    for (int k = 1; k < 10; ++k) mx = fmaxf(mx, lg[k]);
    float s = 0.0f;
    for (int k = 0; k < 10; ++k) s += __expf(lg[k] - mx);
    red[0] = mx;
    red[1] = __logf(s);
  }
  __syncthreads();
  if (threadIdx.x < 10)
    out[g * 10 + threadIdx.x] = lg[threadIdx.x] - red[0] - red[1];
}

// ---------------------------------------------------------------------------
// Orchestration
// ---------------------------------------------------------------------------
extern "C" void kernel_launch(void* const* d_in, const int* in_sizes, int n_in,
                              void* d_out, int out_size, void* d_ws, size_t ws_size,
                              hipStream_t stream) {
  const float* x   = (const float*)d_in[0];
  const float* W1  = (const float*)d_in[1];
  const float* b1  = (const float*)d_in[2];
  const float* W2  = (const float*)d_in[3];
  const float* b2  = (const float*)d_in[4];
  const float* W3  = (const float*)d_in[5];
  const float* b3  = (const float*)d_in[6];
  const float* W4  = (const float*)d_in[7];
  const float* b4  = (const float*)d_in[8];
  const float* Wl  = (const float*)d_in[9];
  const float* bl  = (const float*)d_in[10];
  const int*   ei  = (const int*)d_in[11];   // [2, E] row-major
  const int*   bat = (const int*)d_in[12];
  float* out = (float*)d_out;

  const int N = in_sizes[12];          // N_NODES
  const int E = in_sizes[11] / 2;      // N_EDGES
  const int G = 64;                    // N_GRAPHS
  const int* src = ei;
  const int* dst = ei + E;

  float* ws     = (float*)d_ws;
  float* dinv   = ws;                                 // N
  float* hw     = dinv + N;                           // N*64
  float* x1     = hw   + (size_t)N * HID;             // N*64
  float* x2     = x1   + (size_t)N * HID;             // N*64
  float* x3     = x2   + (size_t)N * HID;             // N*64
  float* pooled = x3   + (size_t)N * HID;             // G*192
  float* cnt    = pooled + (size_t)G * 3 * HID;       // G

  const int NF = N * HID;
  const int TPB = 256;
  dim3 blk(TPB);
  auto cdiv = [](long long a, long long b) { return (unsigned)((a + b - 1) / b); };

  // degree -> dinv
  deg_init<<<cdiv(N, TPB), blk, 0, stream>>>(dinv, N);
  deg_scatter<<<cdiv(E, TPB), blk, 0, stream>>>(dst, dinv, E);
  deg_finalize<<<cdiv(N, TPB), blk, 0, stream>>>(dinv, N);

  // zero pooling buffers (pooled + cnt contiguous)
  zero_f32<<<cdiv(G * 3 * HID + G, TPB), blk, 0, stream>>>(pooled, G * 3 * HID + G);

  const unsigned gGemm = cdiv(N, 128);   // 8 waves x 16 rows per block
  const unsigned gNF   = cdiv(NF, TPB);
  const unsigned gScat = cdiv((long long)E * 16, TPB);

  // Layer 1: x1 = relu(conv(x, W1, b1)), K=128
  gemm_wmma<128><<<gGemm, blk, 0, stream>>>(x, W1, hw, N);
  self_init<<<gNF, blk, 0, stream>>>(hw, dinv, b1, x1, NF);
  edge_scatter<<<gScat, blk, 0, stream>>>(src, dst, dinv, hw, x1, E);
  relu_inplace<<<gNF, blk, 0, stream>>>(x1, NF);

  // Layer 2: x2 = relu(conv(x1, W2, b2)), K=64
  gemm_wmma<64><<<gGemm, blk, 0, stream>>>(x1, W2, hw, N);
  self_init<<<gNF, blk, 0, stream>>>(hw, dinv, b2, x2, NF);
  edge_scatter<<<gScat, blk, 0, stream>>>(src, dst, dinv, hw, x2, E);
  relu_inplace<<<gNF, blk, 0, stream>>>(x2, NF);

  // Layer 3: x3 = relu(conv(x2, W3, b3))
  gemm_wmma<64><<<gGemm, blk, 0, stream>>>(x2, W3, hw, N);
  self_init<<<gNF, blk, 0, stream>>>(hw, dinv, b3, x3, NF);
  edge_scatter<<<gScat, blk, 0, stream>>>(src, dst, dinv, hw, x3, E);
  relu_inplace<<<gNF, blk, 0, stream>>>(x3, NF);

  // Layer 4: x3 = relu(conv(x3, W4, b4))  (in place over x3; hw holds x3@W4)
  gemm_wmma<64><<<gGemm, blk, 0, stream>>>(x3, W4, hw, N);
  self_init<<<gNF, blk, 0, stream>>>(hw, dinv, b4, x3, NF);
  edge_scatter<<<gScat, blk, 0, stream>>>(src, dst, dinv, hw, x3, E);
  relu_inplace<<<gNF, blk, 0, stream>>>(x3, NF);

  // JK-concat mean pool + linear + log_softmax
  pool_sum<<<gNF, blk, 0, stream>>>(x1, x2, x3, bat, pooled, cnt, N);
  classify<<<G, dim3(32), 0, stream>>>(pooled, cnt, Wl, bl, out);
}
